// ETBlock_71614284694160
// MI455X (gfx1250) — matrix operand order repaired and checked by reading, not compiled, not run on hardware
//
#include <hip/hip_runtime.h>
#include <hip/hip_bf16.h>

// ---------------------------------------------------------------------------
// Energy Transformer energy on gfx1250 (MI455X).
//   e = -(1/beta) * sum logsumexp_m(beta * (gWq^T)(gWk^T)^T) - 0.5*sum relu(g Whn^T)^2
// bf16 WMMA (f32 accum) + async global->LDS double-buffered staging + fused
// reductions (scores / h never touch HBM).
// ---------------------------------------------------------------------------

#define B_       2
#define S_       2048
#define D_       1024
#define H_       12
#define QK_      64
#define HN_      4096          // 4*D
#define BS_      (B_ * S_)     // 4096 rows of g
#define HK_      (H_ * QK_)    // 768 projection columns
#define BETA_    0.125f        // 1/sqrt(64)
#define INVBETA_ 8.0f

// LDS tile geometry: K-step = 64 bf16 = 128B per row; pad row stride to 144B
// so half-wave ds_load_b128 hits 16 distinct bank groups (lr*36 mod 64 is a
// permutation of 16 distinct multiples of 4).
#define LDS_STRIDE 144
#define A_ROWS     64
#define B_ROWS     128
#define A_BYTES    (A_ROWS * LDS_STRIDE)   // 9216
#define B_BYTES    (B_ROWS * LDS_STRIDE)   // 18432
#define BUF_BYTES  (A_BYTES + B_BYTES)     // 27648
#define SMEM_BYTES (2 * BUF_BYTES)         // 55296 (< 64KB static, << 320KB/WGP)

typedef __attribute__((ext_vector_type(16))) __bf16 v16bf;
typedef __attribute__((ext_vector_type(8)))  float  v8f;

union Frag16 { v16bf v; uint4 q[2]; };

static __device__ __forceinline__ unsigned short f32_to_bf16(float f) {
    union { float f; unsigned int u; } x; x.f = f;
    unsigned int r = x.u + 0x7FFFu + ((x.u >> 16) & 1u);   // round-to-nearest-even
    return (unsigned short)(r >> 16);
}

static __device__ __forceinline__ v8f wmma_bf16(v16bf a, v16bf b, v8f c) {
    // (neg_a, A, neg_b, B, c_mod, C, reuse_a, reuse_b)
    return __builtin_amdgcn_wmma_f32_16x16x32_bf16(false, a, false, b, (short)0, c, false, false);
}

// ---- async global -> LDS (16B per lane), tracked by ASYNCcnt -------------
static __device__ __forceinline__ void async_b128(unsigned char* lds_ptr,
                                                  const unsigned short* gptr) {
    unsigned lds = (unsigned)(uintptr_t)lds_ptr;   // low 32 bits = LDS offset
    asm volatile("global_load_async_to_lds_b128 %0, %1, off"
                 :: "v"(lds), "v"(gptr) : "memory");
}

// Stage A tile: 64 rows x 64 cols bf16 = 8KB.  512 b128 ops -> 2 per lane.
static __device__ __forceinline__ void stage_tileA(unsigned char* smem, int bufoff,
                                                   const unsigned short* src, int ld,
                                                   int row0, int d0, int tid) {
    const int chunk = tid & 7, row = tid >> 3;          // row 0..31, chunk 0..7
    const unsigned short* g0 = src + (size_t)(row0 + row) * ld + d0 + chunk * 8;
    unsigned char* l0 = smem + bufoff + row * LDS_STRIDE + chunk * 16;
    async_b128(l0, g0);
    async_b128(l0 + 32 * LDS_STRIDE, g0 + (size_t)32 * ld);
}

// Stage B tile: 128 rows x 64 cols bf16 = 16KB.  1024 b128 ops -> 4 per lane.
static __device__ __forceinline__ void stage_tileB(unsigned char* smem, int bufoff,
                                                   const unsigned short* src, int ld,
                                                   int row0, int d0, int tid) {
    const int chunk = tid & 7, row = tid >> 3;
    const unsigned short* g0 = src + (size_t)(row0 + row) * ld + d0 + chunk * 8;
    unsigned char* l0 = smem + bufoff + row * LDS_STRIDE + chunk * 16;
#pragma unroll
    for (int i = 0; i < 4; ++i)
        async_b128(l0 + i * 32 * LDS_STRIDE, g0 + (size_t)(i * 32) * ld);
}

// A fragment (16x32 bf16, MxK) from LDS. soff = 0 or 64 bytes (K sub-step).
// Lane lr(+16*half): row rg*16+lr, K bytes [soff+16*half,+16) and [+32).
static __device__ __forceinline__ v16bf lds_a_frag(const unsigned char* smem, int abase,
                                                   int rg, int soff, int lr, int half) {
    const unsigned char* p = smem + abase + (rg * 16 + lr) * LDS_STRIDE + soff + 16 * half;
    Frag16 f;
    f.q[0] = *(const uint4*)(p);
    f.q[1] = *(const uint4*)(p + 32);
    return f.v;
}

// B fragment (32x16, KxN) = X^T: lane lr holds column cg16*16+lr, K bytes
// [soff+16*half,+16) and [+32) within that X row.
static __device__ __forceinline__ v16bf lds_b_frag(const unsigned char* smem, int bbase,
                                                   int cg16, int soff, int lr, int half) {
    const unsigned char* p = smem + bbase + (cg16 * 16 + lr) * LDS_STRIDE + soff + 16 * half;
    Frag16 f;
    f.q[0] = *(const uint4*)(p);
    f.q[1] = *(const uint4*)(p + 32);
    return f.v;
}

// Direct-from-global fragments (attention GEMM; K=64 tiles stay L2-resident
// and are register double-buffered).
static __device__ __forceinline__ v16bf load_a_frag(const unsigned short* base, int ld,
                                                    int row0, int d0, int lr, int half) {
    const unsigned short* p = base + (size_t)(row0 + lr) * ld + d0 + 8 * half;
    Frag16 f;
    f.q[0] = *(const uint4*)(p);
    f.q[1] = *(const uint4*)(p + 16);
    return f.v;
}
static __device__ __forceinline__ v16bf load_bT_frag(const unsigned short* base, int ld,
                                                     int col0, int d0, int lr, int half) {
    const unsigned short* p = base + (size_t)(col0 + lr) * ld + d0 + 8 * half;
    Frag16 f;
    f.q[0] = *(const uint4*)(p);
    f.q[1] = *(const uint4*)(p + 16);
    return f.v;
}

// ---------------------------------------------------------------------------
// Kernel 0: fp32 -> bf16 conversion
// ---------------------------------------------------------------------------
__global__ void et_cvt_bf16(const float* __restrict__ in, unsigned short* __restrict__ out, int n) {
    for (int i = blockIdx.x * blockDim.x + threadIdx.x; i < n; i += gridDim.x * blockDim.x)
        out[i] = f32_to_bf16(in[i]);
}

// ---------------------------------------------------------------------------
// Shared tiled-GEMM body: workgroup tile 64 rows x 128 cols, K-step 64,
// async-LDS double-buffered.  8 waves: wave -> 16-row x 64-col strip.
// Per barrier interval: 6 async b128/wave staged, 20 ds_load_b128 batched,
// then 8 back-to-back WMMAs.
// ---------------------------------------------------------------------------
static __device__ __forceinline__ void gemm_tile_64x128(
        unsigned char* smem, v8f acc[4],
        const unsigned short* __restrict__ Asrc, const unsigned short* __restrict__ Bsrc,
        int row0, int col0, int tid, int wr, int wc, int lr, int half) {

    stage_tileA(smem, 0,       Asrc, D_, row0, 0, tid);
    stage_tileB(smem, A_BYTES, Bsrc, D_, col0, 0, tid);

    for (int d0 = 0; d0 < D_; d0 += 64) {
        const int cur = (d0 >> 6) & 1, nxt = cur ^ 1;
        if (d0 + 64 < D_) {
            stage_tileA(smem, nxt * BUF_BYTES,           Asrc, D_, row0, d0 + 64, tid);
            stage_tileB(smem, nxt * BUF_BYTES + A_BYTES, Bsrc, D_, col0, d0 + 64, tid);
            // async ops complete in order: <=6 outstanding => the 6 older
            // (current buffer) ops are done.
            asm volatile("s_wait_asynccnt 0x6" ::: "memory");
        } else {
            asm volatile("s_wait_asynccnt 0x0" ::: "memory");
        }
        __syncthreads();

        const int ab = cur * BUF_BYTES, bb = ab + A_BYTES;
        // Request every fragment first so ds_loads batch under one wait,
        // then run the 8 WMMAs without intervening LDS stalls.
        v16bf a0 = lds_a_frag(smem, ab, wr, 0,  lr, half);
        v16bf a1 = lds_a_frag(smem, ab, wr, 64, lr, half);
        v16bf bf[8];
#pragma unroll
        for (int j = 0; j < 4; ++j) {
            bf[2 * j]     = lds_b_frag(smem, bb, wc * 4 + j, 0,  lr, half);
            bf[2 * j + 1] = lds_b_frag(smem, bb, wc * 4 + j, 64, lr, half);
        }
#pragma unroll
        for (int j = 0; j < 4; ++j) {
            acc[j] = wmma_bf16(a0, bf[2 * j],     acc[j]);
            acc[j] = wmma_bf16(a1, bf[2 * j + 1], acc[j]);
        }
        __syncthreads();   // all waves done reading before buffer is restaged
    }
}

// ---------------------------------------------------------------------------
// Kernel 1: q = g Wq^T, k = g Wk^T.
// grid: 2 matrices * 64 row-groups * 6 col-groups = 768 blocks.
// ---------------------------------------------------------------------------
__global__ void et_proj(const unsigned short* __restrict__ gbf,
                        const unsigned short* __restrict__ wqbf,
                        const unsigned short* __restrict__ wkbf,
                        unsigned short* __restrict__ qbf,
                        unsigned short* __restrict__ kbf) {
    __shared__ __align__(16) unsigned char smem[SMEM_BYTES];
    const int tid  = threadIdx.x;
    const int wave = tid >> 5, lane = tid & 31, lr = lane & 15, half = lane >> 4;
    const int wr = wave >> 1, wc = wave & 1;

    int bid = blockIdx.x;
    const int which = bid >= 384 ? 1 : 0;
    bid -= which * 384;
    const int rt = bid / 6;             // 64-row group of g
    const int cg = bid % 6;             // 128-col group of W
    const int row0 = rt * 64, col0 = cg * 128;

    const unsigned short* W   = which ? wkbf : wqbf;
    unsigned short*       dst = which ? kbf  : qbf;

    v8f acc[4];
#pragma unroll
    for (int j = 0; j < 4; ++j)
#pragma unroll
        for (int i = 0; i < 8; ++i) acc[j][i] = 0.0f;

    gemm_tile_64x128(smem, acc, gbf, W, row0, col0, tid, wr, wc, lr, half);

    // C layout: VGPR r -> row (r + 8*half), column lr
#pragma unroll
    for (int j = 0; j < 4; ++j) {
        const int col = col0 + wc * 64 + j * 16 + lr;   // 0..767
        const int h   = col >> 6;
        const int kc  = col & 63;
#pragma unroll
        for (int r = 0; r < 8; ++r) {
            const int nrow = row0 + wr * 16 + r + 8 * half;  // 0..4095
            const int b    = nrow >> 11;
            const int n    = nrow & (S_ - 1);
            size_t idx = (((size_t)(b * H_ + h)) * S_ + n) * QK_ + kc;
            dst[idx] = f32_to_bf16(acc[j][r]);
        }
    }
}

// ---------------------------------------------------------------------------
// Kernel 2: e_attn, flash-style online logsumexp.  One wave per (b,h,16-query
// tile); q fragments resident, key B-fragments register double-buffered.
// jobs: 24 * 128 = 3072 waves (384 blocks).
// ---------------------------------------------------------------------------
__global__ void et_attn(const unsigned short* __restrict__ qbf,
                        const unsigned short* __restrict__ kbf,
                        float* __restrict__ out) {
    const int wid  = (blockIdx.x * blockDim.x + threadIdx.x) >> 5;
    const int lane = threadIdx.x & 31, lr = lane & 15, half = lane >> 4;
    const int bh = wid >> 7;        // 0..23
    const int nt = wid & 127;       // query tile

    const unsigned short* qb = qbf + (size_t)bh * S_ * QK_;
    const unsigned short* kb = kbf + (size_t)bh * S_ * QK_;

    const v16bf a0 = load_a_frag(qb, QK_, nt * 16, 0,  lr, half);
    const v16bf a1 = load_a_frag(qb, QK_, nt * 16, 32, lr, half);

    float rmax[8], rsum[8];
#pragma unroll
    for (int r = 0; r < 8; ++r) { rmax[r] = -INFINITY; rsum[r] = 0.0f; }

    v16bf b0 = load_bT_frag(kb, QK_, 0, 0,  lr, half);
    v16bf b1 = load_bT_frag(kb, QK_, 0, 32, lr, half);

    for (int mt = 0; mt < 128; ++mt) {
        v16bf nb0 = b0, nb1 = b1;
        if (mt + 1 < 128) {   // prefetch next key tile while this one computes
            nb0 = load_bT_frag(kb, QK_, (mt + 1) * 16, 0,  lr, half);
            nb1 = load_bT_frag(kb, QK_, (mt + 1) * 16, 32, lr, half);
        }
        v8f c;
#pragma unroll
        for (int i = 0; i < 8; ++i) c[i] = 0.0f;
        c = wmma_bf16(a0, b0, c);
        c = wmma_bf16(a1, b1, c);

#pragma unroll
        for (int r = 0; r < 8; ++r) {
            float s = BETA_ * c[r];
            float tmax = s;
#pragma unroll
            for (int m = 1; m < 16; m <<= 1) tmax = fmaxf(tmax, __shfl_xor(tmax, m, 16));
            const float nmax = fmaxf(rmax[r], tmax);
            float e = __expf(s - nmax);
#pragma unroll
            for (int m = 1; m < 16; m <<= 1) e += __shfl_xor(e, m, 16);
            rsum[r] = rsum[r] * __expf(rmax[r] - nmax) + e;
            rmax[r] = nmax;
        }
        b0 = nb0; b1 = nb1;
    }

    float part = 0.0f;
#pragma unroll
    for (int r = 0; r < 8; ++r) part += rmax[r] + __logf(rsum[r]);
    // half 0 holds rows 0..7, half 1 rows 8..15
    if (lr == 0) atomicAdd(out, -INVBETA_ * part);
}

// ---------------------------------------------------------------------------
// Kernel 3: e_hn = -0.5*sum relu(g Whn^T)^2, fused relu^2 epilogue.
// grid: 64 row-groups * 32 col-groups = 2048 blocks.
// ---------------------------------------------------------------------------
__global__ void et_hop(const unsigned short* __restrict__ gbf,
                       const unsigned short* __restrict__ whnbf,
                       float* __restrict__ out) {
    __shared__ __align__(16) unsigned char smem[SMEM_BYTES];
    const int tid  = threadIdx.x;
    const int wave = tid >> 5, lane = tid & 31, lr = lane & 15, half = lane >> 4;
    const int wr = wave >> 1, wc = wave & 1;

    const int rt = blockIdx.x >> 5;    // 0..63
    const int cg = blockIdx.x & 31;    // 0..31
    const int row0 = rt * 64, col0 = cg * 128;

    v8f acc[4];
#pragma unroll
    for (int j = 0; j < 4; ++j)
#pragma unroll
        for (int i = 0; i < 8; ++i) acc[j][i] = 0.0f;

    gemm_tile_64x128(smem, acc, gbf, whnbf, row0, col0, tid, wr, wc, lr, half);

    float s = 0.0f;
#pragma unroll
    for (int j = 0; j < 4; ++j)
#pragma unroll
        for (int r = 0; r < 8; ++r) {
            float v = fmaxf(acc[j][r], 0.0f);
            s += 0.5f * v * v;
        }
#pragma unroll
    for (int m = 1; m < 32; m <<= 1) s += __shfl_xor(s, m, 32);
    if (lane == 0) atomicAdd(out, -s);
}

// ---------------------------------------------------------------------------
extern "C" void kernel_launch(void* const* d_in, const int* in_sizes, int n_in,
                              void* d_out, int out_size, void* d_ws, size_t ws_size,
                              hipStream_t stream) {
    (void)in_sizes; (void)n_in; (void)out_size; (void)ws_size;

    const float* g   = (const float*)d_in[0];   // [2,2048,1024]
    const float* Wq  = (const float*)d_in[1];   // [12,64,1024]
    const float* Wk  = (const float*)d_in[2];   // [12,64,1024]
    const float* Whn = (const float*)d_in[3];   // [4096,1024]
    float* out = (float*)d_out;

    // workspace layout (bf16 = unsigned short), total ~31 MiB (L2-resident)
    unsigned short* gbf   = (unsigned short*)d_ws;
    unsigned short* wqbf  = gbf   + (size_t)BS_ * D_;
    unsigned short* wkbf  = wqbf  + (size_t)HK_ * D_;
    unsigned short* whnbf = wkbf  + (size_t)HK_ * D_;
    unsigned short* qbf   = whnbf + (size_t)HN_ * D_;
    unsigned short* kbf   = qbf   + (size_t)B_ * H_ * S_ * QK_;

    hipMemsetAsync(d_out, 0, sizeof(float), stream);

    const int ng = BS_ * D_, nw = HK_ * D_, nh = HN_ * D_;
    et_cvt_bf16<<<(ng + 255) / 256, 256, 0, stream>>>(g,   gbf,   ng);
    et_cvt_bf16<<<(nw + 255) / 256, 256, 0, stream>>>(Wq,  wqbf,  nw);
    et_cvt_bf16<<<(nw + 255) / 256, 256, 0, stream>>>(Wk,  wkbf,  nw);
    et_cvt_bf16<<<(nh + 255) / 256, 256, 0, stream>>>(Whn, whnbf, nh);

    et_proj<<<768,  256, 0, stream>>>(gbf, wqbf, wkbf, qbf, kbf);
    et_attn<<<384,  256, 0, stream>>>(qbf, kbf, out);
    et_hop <<<2048, 256, 0, stream>>>(gbf, whnbf, out);
}
